// DiT_33449205301772
// MI455X (gfx1250) — compile-verified, matrix operand order checked by reading, not run on hardware
//
#include <hip/hip_runtime.h>
#include <hip/hip_bf16.h>

// ---------------- problem constants ----------------
#define B_     4
#define T_     256
#define L_     1024      // T * DMAX
#define ALD_   128
#define U_     512
#define MEL_   100
#define KPAD_  128       // MEL padded to K/N tile size
#define HEADS_ 8
#define HD_    64
#define FF_    2048
#define DEPTH_ 6
#define SIGMA_ 1e-5f

typedef __attribute__((ext_vector_type(16))) __bf16 v16bf;
typedef __attribute__((ext_vector_type(8)))  __bf16 v8bf;
typedef __attribute__((ext_vector_type(8)))  float  v8f;

__device__ __forceinline__ unsigned pack2bf(float a, float b)
{
    union { __bf16 h[2]; unsigned u; } u2;
    u2.h[0] = (__bf16)a; u2.h[1] = (__bf16)b;
    return u2.u;
}

__device__ __forceinline__ v16bf ld_frag(const __bf16* p)   // p 16B-aligned
{
    v8bf lo = *reinterpret_cast<const v8bf*>(p);
    v8bf hi = *reinterpret_cast<const v8bf*>(p + 8);
    return __builtin_shufflevector(lo, hi, 0,1,2,3,4,5,6,7,8,9,10,11,12,13,14,15);
}

// =====================================================================
// Full-tile GEMM: Cout[M,N] = act(A[M,K] @ W[K,N] + bias) + Cin
// Requires M%64==0, N%128==0, K%32==0 (guaranteed by padding).
// Block = 128 thr = 4 waves; block tile 64x128; wave tile 32x64
// -> 8 x v_wmma_f32_16x16x32_bf16 per K-step per wave.
// act: 0 none, 1 gelu(tanh), 2 silu
// =====================================================================
__global__ __launch_bounds__(128)
void gemm_bf16_wmma(const float* __restrict__ A, const float* __restrict__ W,
                    const float* __restrict__ bias, const float* __restrict__ Cin,
                    float* __restrict__ Cout, int M, int N, int K, int act)
{
    __shared__ __bf16 As [64][40];    // A rows   : As[m][k]   (80B stride, 16B aligned)
    __shared__ __bf16 BsT[128][40];   // W columns: BsT[n][k]

    const int tid  = threadIdx.x;
    const int lane = tid & 31;
    const int wave = tid >> 5;
    const int m0   = blockIdx.y * 64;
    const int n0   = blockIdx.x * 128;
    const int wm   = (wave >> 1) * 32;
    const int wn   = (wave & 1) * 64;

    v8f acc[2][4] = {};

    for (int k0 = 0; k0 < K; k0 += 32) {
        #pragma unroll
        for (int i = 0; i < 4; ++i) {
            int idx = tid + i * 128;
            int r = idx >> 3, c4 = (idx & 7) << 2;
            const float4 a = *reinterpret_cast<const float4*>(&A[(long)(m0 + r) * K + k0 + c4]);
            uint2 pk; pk.x = pack2bf(a.x, a.y); pk.y = pack2bf(a.z, a.w);
            *reinterpret_cast<uint2*>(&As[r][c4]) = pk;
        }
        #pragma unroll
        for (int i = 0; i < 8; ++i) {
            int idx = tid + i * 128;
            int col = idx & 127, kc = (idx >> 7) << 2;
            const float* wp = &W[(long)(k0 + kc) * N + n0 + col];
            float w0 = wp[0], w1 = wp[N], w2 = wp[2 * N], w3 = wp[3 * N];
            uint2 pk; pk.x = pack2bf(w0, w1); pk.y = pack2bf(w2, w3);
            *reinterpret_cast<uint2*>(&BsT[col][kc]) = pk;
        }
        __syncthreads();

        if (k0 + 32 < K) {
            __builtin_prefetch(&A[(long)(m0 + (tid >> 1)) * K + k0 + 32]);
            __builtin_prefetch(&W[(long)(k0 + 32 + (tid >> 2)) * N + n0]);
        }

        const int kb = (lane >> 4) << 4;
        v16bf afr[2], bfc[4];
        #pragma unroll
        for (int r = 0; r < 2; ++r)
            afr[r] = ld_frag(&As[wm + r * 16 + (lane & 15)][kb]);
        #pragma unroll
        for (int c = 0; c < 4; ++c)
            bfc[c] = ld_frag(&BsT[wn + c * 16 + (lane & 15)][kb]);

        #pragma unroll
        for (int r = 0; r < 2; ++r)
            #pragma unroll
            for (int c = 0; c < 4; ++c)
                acc[r][c] = __builtin_amdgcn_wmma_f32_16x16x32_bf16(
                    false, afr[r], false, bfc[c], (short)0, acc[r][c], false, false);

        __syncthreads();
    }

    #pragma unroll
    for (int r = 0; r < 2; ++r)
        #pragma unroll
        for (int c = 0; c < 4; ++c) {
            const int n  = n0 + wn + c * 16 + (lane & 15);
            const int mb = m0 + wm + r * 16 + ((lane >> 4) << 3);
            #pragma unroll
            for (int i = 0; i < 8; ++i) {
                const int m = mb + i;
                float v = acc[r][c][i];
                if (bias) v += bias[n];
                if (act == 1) {
                    float x3 = v * v * v;
                    v = 0.5f * v * (1.0f + tanhf(0.7978845608f * (v + 0.044715f * x3)));
                } else if (act == 2) {
                    v = v / (1.0f + __expf(-v));
                }
                if (Cin) v += Cin[(long)m * N + n];
                Cout[(long)m * N + n] = v;
            }
        }
}

// =====================================================================
// LayerNorm over last dim (N); one block per row. In-place safe.
// =====================================================================
__global__ __launch_bounds__(256)
void layernorm_kernel(const float* __restrict__ x, const float* __restrict__ g,
                      const float* __restrict__ b, float* __restrict__ y, int N)
{
    __shared__ float red[256];
    const int row = blockIdx.x, tid = threadIdx.x;
    const float* xr = x + (long)row * N;

    float s = 0.f;
    for (int i = tid; i < N; i += 256) s += xr[i];
    red[tid] = s; __syncthreads();
    for (int o = 128; o > 0; o >>= 1) { if (tid < o) red[tid] += red[tid + o]; __syncthreads(); }
    float mean = red[0] / N;
    __syncthreads();

    float v = 0.f;
    for (int i = tid; i < N; i += 256) { float d = xr[i] - mean; v += d * d; }
    red[tid] = v; __syncthreads();
    for (int o = 128; o > 0; o >>= 1) { if (tid < o) red[tid] += red[tid + o]; __syncthreads(); }
    float inv = rsqrtf(red[0] / N + 1e-5f);

    for (int i = tid; i < N; i += 256)
        y[(long)row * N + i] = (xr[i] - mean) * inv * g[i] + b[i];
}

// =====================================================================
// Length regulation: cumsum + searchsorted(right) per batch
// =====================================================================
__global__ void lr_scan_kernel(const int* __restrict__ dur,
                               const unsigned char* __restrict__ pmask,
                               int* __restrict__ tok, int* __restrict__ keep,
                               int* __restrict__ validk)
{
    __shared__ int cum[T_];
    const int b = blockIdx.x, tid = threadIdx.x;
    if (tid == 0) {
        int s = 0;
        for (int i = 0; i < T_; ++i) { int e = dur[b * T_ + i]; if (e < 1) e = 1; s += e; cum[i] = s; }
    }
    __syncthreads();
    const int total = cum[T_ - 1];
    for (int f = tid; f < L_; f += 256) {
        int lo = 0, hi = T_;
        while (lo < hi) { int mid = (lo + hi) >> 1; if (cum[mid] <= f) lo = mid + 1; else hi = mid; }
        int t = lo > T_ - 1 ? T_ - 1 : lo;
        int valid = (f < total);
        int kp = valid && (dur[b * T_ + t] > 0);
        tok[b * L_ + f]    = t;
        keep[b * L_ + f]   = kp;
        validk[b * L_ + f] = kp && (pmask[b * T_ + t] == 0);
    }
}

__global__ void lr_gather_kernel(const float* __restrict__ ctx, const int* __restrict__ tok,
                                 const int* __restrict__ keep, float* __restrict__ out)
{
    long i = (long)blockIdx.x * 256 + threadIdx.x;
    if (i >= (long)B_ * L_ * ALD_) return;
    int d = i % ALD_;
    long f = i / ALD_;
    int b = (int)(f / L_);
    int t = tok[f];
    out[i] = keep[f] ? ctx[((long)b * T_ + t) * ALD_ + d] : 0.0f;
}

// =====================================================================
// w (stride KPAD_, zero-padded cols) and v_target (stride MEL_)
// =====================================================================
__global__ void prep_wv_kernel(const float* __restrict__ target, const float* __restrict__ noise,
                               const float* __restrict__ times, float* __restrict__ w,
                               float* __restrict__ vt)
{
    long i = (long)blockIdx.x * 256 + threadIdx.x;
    if (i >= (long)B_ * L_ * KPAD_) return;
    long row = i >> 7;
    int  c   = i & (KPAD_ - 1);
    if (c < MEL_) {
        int b = (int)(row / L_);
        float t = times[b];
        long src = row * MEL_ + c;
        float tg = target[src], nz = noise[src];
        w[i] = (1.0f - (1.0f - SIGMA_) * t) * nz + t * tg;
        vt[src] = tg - (1.0f - SIGMA_) * nz;
    } else {
        w[i] = 0.0f;
    }
}

// ---- zero-padded weight copies ----
__global__ void pad_wnp_kernel(const float* __restrict__ w_np, float* __restrict__ out)
{
    int i = blockIdx.x * 256 + threadIdx.x;        // KPAD_ x U_
    if (i >= KPAD_ * U_) return;
    int r = i / U_;
    out[i] = (r < MEL_) ? w_np[(long)(U_ + r) * U_ + (i % U_)] : 0.0f;
}

__global__ void pad_wout_kernel(const float* __restrict__ w_out, const float* __restrict__ b_out,
                                float* __restrict__ wpad, float* __restrict__ bpad)
{
    int i = blockIdx.x * 256 + threadIdx.x;        // U_ x KPAD_
    if (i < U_ * KPAD_) {
        int r = i >> 7, c = i & (KPAD_ - 1);
        wpad[i] = (c < MEL_) ? w_out[r * MEL_ + c] : 0.0f;
    }
    if (i < KPAD_) bpad[i] = (i < MEL_) ? b_out[i] : 0.0f;
}

// time embedding, rows padded to 64 (rows >= B_ are zero)
__global__ void temb_raw_kernel(const float* __restrict__ times, float* __restrict__ out)
{
    const int b = blockIdx.x, tid = threadIdx.x;
    const int half = U_ / 2;
    if (b >= B_) {
        for (int i = tid; i < U_; i += 256) out[b * U_ + i] = 0.0f;
        return;
    }
    const float t = times[b];
    for (int i = tid; i < U_; i += 256) {
        int j = (i < half) ? i : i - half;
        float fr = __expf(-logf(10000.0f) * (float)j / (float)half);
        out[b * U_ + i] = (i < half) ? cosf(t * fr) : sinf(t * fr);
    }
}

__global__ void add_temb_kernel(float* __restrict__ x, const float* __restrict__ temb)
{
    long i = (long)blockIdx.x * 256 + threadIdx.x;
    if (i >= (long)B_ * L_ * U_) return;
    int u = i % U_;
    int b = (int)(i / ((long)L_ * U_));
    x[i] += temb[b * U_ + u];
}

// =====================================================================
// Flash attention with WMMA.
// Block = 256 thr = 8 waves; block -> (b, h, 128 query rows); wave -> 16 rows.
// Keys/values staged 64 at a time (K row-major, V transposed) as bf16.
// S = Q K^T: 2 WMMA per 16x16 S tile (HD=64). Softmax in C-layout with
// 16-lane-group shuffles. P bounced through per-wave LDS -> A-frags for
// O += P V (4 d-tiles per 32-key step). 16 WMMA per wave per 64-key block.
// =====================================================================
__global__ __launch_bounds__(256)
void attn_kernel(const float* __restrict__ qkv, const int* __restrict__ validk,
                 float* __restrict__ out)
{
    __shared__ __bf16 Kt [64][72];       // K rows  [key][d]   (144B stride)
    __shared__ __bf16 VtT[64][72];       // V cols  [d][key]
    __shared__ __bf16 Pw [8][16][40];    // per-wave P scratch (80B stride)
    __shared__ int    vm[64];

    const int blk  = blockIdx.x;                 // B*HEADS*(L/128)
    const int qb   = blk % (L_ / 128);
    const int h    = (blk / (L_ / 128)) % HEADS_;
    const int b    = blk / ((L_ / 128) * HEADS_);
    const int tid  = threadIdx.x;
    const int wave = tid >> 5;
    const int lane = tid & 31;
    const int qbase = qb * 128;
    const int q0    = qbase + wave * 16;         // this wave's 16 query rows
    const int hi16  = (lane >> 4);               // 0 or 1
    const long RS   = 3 * U_;                    // qkv row stride

    // ---- load Q fragments for both K-steps (held in regs) ----
    v16bf qf[2];
    {
        const float* qp = qkv + ((long)(b * L_ + q0 + (lane & 15))) * RS + h * HD_;
        #pragma unroll
        for (int s2 = 0; s2 < 2; ++s2) {
            union { v16bf v; uint2 u2[4]; } tmp;
            const int dbase = s2 * 32 + (hi16 << 4);
            #pragma unroll
            for (int j = 0; j < 4; ++j) {
                const float4 f = *reinterpret_cast<const float4*>(qp + dbase + j * 4);
                uint2 pk; pk.x = pack2bf(f.x, f.y); pk.y = pack2bf(f.z, f.w);
                tmp.u2[j] = pk;
            }
            qf[s2] = tmp.v;
        }
    }

    v8f  oacc[4] = {};
    float mrun[8], lrun[8];
    #pragma unroll
    for (int r = 0; r < 8; ++r) { mrun[r] = -3.0e38f; lrun[r] = 0.f; }

    for (int kb = 0; kb <= qbase + 127; kb += 64) {     // uniform over block
        // ---- stage K (row-major) and V (transposed) as bf16 ----
        #pragma unroll
        for (int i = 0; i < 4; ++i) {
            int idx = tid + i * 256;                    // 1024 float4 chunks
            int key = idx >> 4, d4 = (idx & 15) << 2;
            const float4 f = *reinterpret_cast<const float4*>(
                qkv + ((long)(b * L_ + kb + key)) * RS + U_ + h * HD_ + d4);
            uint2 pk; pk.x = pack2bf(f.x, f.y); pk.y = pack2bf(f.z, f.w);
            *reinterpret_cast<uint2*>(&Kt[key][d4]) = pk;
        }
        #pragma unroll
        for (int i = 0; i < 4; ++i) {
            int idx = tid + i * 256;
            int d = idx & 63, k4 = (idx >> 6) << 2;
            const float* vp = qkv + ((long)(b * L_ + kb + k4)) * RS + 2 * U_ + h * HD_ + d;
            float v0 = vp[0], v1 = vp[RS], v2 = vp[2 * RS], v3 = vp[3 * RS];
            uint2 pk; pk.x = pack2bf(v0, v1); pk.y = pack2bf(v2, v3);
            *reinterpret_cast<uint2*>(&VtT[d][k4]) = pk;
        }
        if (tid < 64) vm[tid] = validk[b * L_ + kb + tid];
        __syncthreads();

        if (q0 + 15 >= kb) {                            // wave-uniform work check
            // ---- S tiles: 4 key subtiles x (2 WMMA over d) ----
            v8f st[4] = {};
            #pragma unroll
            for (int t = 0; t < 4; ++t) {
                #pragma unroll
                for (int s2 = 0; s2 < 2; ++s2) {
                    v16bf kf = ld_frag(&Kt[t * 16 + (lane & 15)][s2 * 32 + (hi16 << 4)]);
                    st[t] = __builtin_amdgcn_wmma_f32_16x16x32_bf16(
                        false, qf[s2], false, kf, (short)0, st[t], false, false);
                }
            }
            // ---- scale + causal/key mask (C layout) ----
            #pragma unroll
            for (int t = 0; t < 4; ++t) {
                const int key = kb + t * 16 + (lane & 15);
                const bool kv = vm[t * 16 + (lane & 15)] != 0;
                #pragma unroll
                for (int r = 0; r < 8; ++r) {
                    const int qrow = q0 + (hi16 << 3) + r;
                    const bool ok = (key <= qrow) && kv;
                    st[t][r] = ok ? st[t][r] * 0.125f : -1.0e9f;
                }
            }
            // ---- online softmax stats (row = 16-lane group reduce) ----
            float scl[8];
            #pragma unroll
            for (int r = 0; r < 8; ++r) {
                float tm = fmaxf(fmaxf(st[0][r], st[1][r]), fmaxf(st[2][r], st[3][r]));
                #pragma unroll
                for (int o2 = 1; o2 < 16; o2 <<= 1) tm = fmaxf(tm, __shfl_xor(tm, o2, 32));
                float mn = fmaxf(mrun[r], tm);
                scl[r] = __expf(mrun[r] - mn);
                mrun[r] = mn;
            }
            #pragma unroll
            for (int t = 0; t < 4; ++t)
                #pragma unroll
                for (int r = 0; r < 8; ++r)
                    st[t][r] = __expf(st[t][r] - mrun[r]);
            #pragma unroll
            for (int r = 0; r < 8; ++r) {
                float rs = st[0][r] + st[1][r] + st[2][r] + st[3][r];
                #pragma unroll
                for (int o2 = 1; o2 < 16; o2 <<= 1) rs += __shfl_xor(rs, o2, 32);
                lrun[r] = lrun[r] * scl[r] + rs;
                #pragma unroll
                for (int dt = 0; dt < 4; ++dt) oacc[dt][r] = oacc[dt][r] * scl[r];
            }
            // ---- O += P V in two 32-key steps (P via per-wave LDS) ----
            #pragma unroll
            for (int ks = 0; ks < 2; ++ks) {
                #pragma unroll
                for (int th = 0; th < 2; ++th) {
                    const int t = 2 * ks + th;
                    #pragma unroll
                    for (int r = 0; r < 8; ++r)
                        Pw[wave][(hi16 << 3) + r][th * 16 + (lane & 15)] = (__bf16)st[t][r];
                }
                v16bf pf = ld_frag(&Pw[wave][lane & 15][hi16 << 4]);
                #pragma unroll
                for (int dt = 0; dt < 4; ++dt) {
                    v16bf vf = ld_frag(&VtT[dt * 16 + (lane & 15)][ks * 32 + (hi16 << 4)]);
                    oacc[dt] = __builtin_amdgcn_wmma_f32_16x16x32_bf16(
                        false, pf, false, vf, (short)0, oacc[dt], false, false);
                }
            }
        }
        __syncthreads();
    }

    // ---- write O (heads interleaved) ----
    #pragma unroll
    for (int dt = 0; dt < 4; ++dt) {
        const int d = dt * 16 + (lane & 15);
        #pragma unroll
        for (int r = 0; r < 8; ++r) {
            const int row = q0 + (hi16 << 3) + r;
            out[((long)(b * L_ + row)) * U_ + h * HD_ + d] = oacc[dt][r] / lrun[r];
        }
    }
}

// =====================================================================
// Deterministic two-stage masked-MSE reduction (pred has stride KPAD_)
// =====================================================================
__global__ __launch_bounds__(256)
void loss_partial_kernel(const float* __restrict__ v, const float* __restrict__ vt,
                         const int* __restrict__ validk, float* __restrict__ partial)
{
    __shared__ float red[256];
    long i = (long)blockIdx.x * 256 + threadIdx.x;
    float s = 0.f;
    if (i < (long)B_ * L_ * MEL_) {
        long row = i / MEL_;
        int  col = (int)(i % MEL_);
        if (validk[row]) {
            float d = v[row * KPAD_ + col] - vt[i];
            s = d * d;
        }
    }
    red[threadIdx.x] = s; __syncthreads();
    for (int o = 128; o; o >>= 1) { if (threadIdx.x < o) red[threadIdx.x] += red[threadIdx.x + o]; __syncthreads(); }
    if (threadIdx.x == 0) partial[blockIdx.x] = red[0];
}

__global__ __launch_bounds__(256)
void loss_final_kernel(const float* __restrict__ partial, int nparts,
                       const int* __restrict__ validk, float* __restrict__ out)
{
    __shared__ float red[256], redn[256];
    const int tid = threadIdx.x;
    float s = 0.f, n = 0.f;
    for (int i = tid; i < nparts; i += 256) s += partial[i];
    for (int i = tid; i < B_ * L_; i += 256) n += (float)validk[i];
    red[tid] = s; redn[tid] = n; __syncthreads();
    for (int o = 128; o; o >>= 1) {
        if (tid < o) { red[tid] += red[tid + o]; redn[tid] += redn[tid + o]; }
        __syncthreads();
    }
    if (tid == 0) out[0] = red[0] / (fmaxf(redn[0], 1.0f) * (float)MEL_);
}

// =====================================================================
// Host driver
// =====================================================================
extern "C" void kernel_launch(void* const* d_in, const int* in_sizes, int n_in,
                              void* d_out, int out_size, void* d_ws, size_t ws_size,
                              hipStream_t stream)
{
    const float*         target = (const float*)d_in[0];
    const unsigned char* pmask  = (const unsigned char*)d_in[1];   // bool
    const float*         ctx    = (const float*)d_in[2];
    const int*           dur    = (const int*)d_in[3];
    const float*         times  = (const float*)d_in[4];
    const float*         noise  = (const float*)d_in[5];
    int p = 6;
    const float* w_ctx   = (const float*)d_in[p++];
    const float* b_ctx   = (const float*)d_in[p++];
    const float* ln_ctx_g= (const float*)d_in[p++];
    const float* ln_ctx_b= (const float*)d_in[p++];
    const float* w_np    = (const float*)d_in[p++];   // (U+MEL, U)
    const float* b_np    = (const float*)d_in[p++];
    const float* ln_np_g = (const float*)d_in[p++];
    const float* ln_np_b = (const float*)d_in[p++];
    const float* t_w1    = (const float*)d_in[p++];
    const float* t_b1    = (const float*)d_in[p++];
    const float* t_w2    = (const float*)d_in[p++];
    const float* t_b2    = (const float*)d_in[p++];
    const int layer_base = p;                         // 6 layers x 12 tensors
    p += DEPTH_ * 12;
    const float* ln_f_g  = (const float*)d_in[p++];
    const float* ln_f_b  = (const float*)d_in[p++];
    const float* w_out   = (const float*)d_in[p++];
    const float* b_out   = (const float*)d_in[p++];
    (void)in_sizes; (void)n_in; (void)out_size; (void)ws_size;

    // ---- workspace carve-out ----
    char* ws = (char*)d_ws;
    size_t off = 0;
    auto alloc = [&](size_t bytes) -> void* {
        void* ptr = (void*)(ws + off);
        off += (bytes + 255) & ~(size_t)255;
        return ptr;
    };
    int*   tok     = (int*)  alloc(sizeof(int)   * B_ * L_);
    int*   keep    = (int*)  alloc(sizeof(int)   * B_ * L_);
    int*   validk  = (int*)  alloc(sizeof(int)   * B_ * L_);
    float* ctxf    = (float*)alloc(sizeof(float) * (long)B_ * L_ * ALD_);
    float* xbuf    = (float*)alloc(sizeof(float) * (long)B_ * L_ * U_);
    float* hbuf    = (float*)alloc(sizeof(float) * (long)B_ * L_ * U_);
    float* qkvbuf  = (float*)alloc(sizeof(float) * (long)B_ * L_ * 3 * U_);
    float* obuf    = (float*)alloc(sizeof(float) * (long)B_ * L_ * U_);
    float* ffbuf   = (float*)alloc(sizeof(float) * (long)B_ * L_ * FF_);
    float* wbuf    = (float*)alloc(sizeof(float) * (long)B_ * L_ * KPAD_);  // zero-padded
    float* vtbuf   = (float*)alloc(sizeof(float) * (long)B_ * L_ * MEL_);
    float* wnp_pad = (float*)alloc(sizeof(float) * KPAD_ * U_);
    float* wout_pad= (float*)alloc(sizeof(float) * U_ * KPAD_);
    float* bout_pad= (float*)alloc(sizeof(float) * KPAD_);
    float* traw    = (float*)alloc(sizeof(float) * 64 * U_);    // M padded to 64
    float* thid    = (float*)alloc(sizeof(float) * 64 * U_);
    float* temb    = (float*)alloc(sizeof(float) * 64 * U_);
    float* outbuf  = (float*)alloc(sizeof(float) * (long)B_ * L_ * KPAD_);
    const int nparts = (B_ * L_ * MEL_ + 255) / 256;
    float* partial = (float*)alloc(sizeof(float) * nparts);

    const int ML = B_ * L_;                     // 4096 rows

    // ---- padding / preprocessing ----
    pad_wnp_kernel <<<(KPAD_ * U_ + 255) / 256, 256, 0, stream>>>(w_np, wnp_pad);
    pad_wout_kernel<<<(U_ * KPAD_ + 255) / 256, 256, 0, stream>>>(w_out, b_out, wout_pad, bout_pad);
    lr_scan_kernel<<<B_, 256, 0, stream>>>(dur, pmask, tok, keep, validk);
    lr_gather_kernel<<<(B_ * L_ * ALD_ + 255) / 256, 256, 0, stream>>>(ctx, tok, keep, ctxf);
    prep_wv_kernel<<<(B_ * L_ * KPAD_ + 255) / 256, 256, 0, stream>>>(target, noise, times, wbuf, vtbuf);

    // ---- context projection + LN ----
    dim3 gU(U_ / 128, ML / 64);
    gemm_bf16_wmma<<<gU, 128, 0, stream>>>(ctxf, w_ctx, b_ctx, nullptr, hbuf, ML, U_, ALD_, 0);
    layernorm_kernel<<<ML, 256, 0, stream>>>(hbuf, ln_ctx_g, ln_ctx_b, hbuf, U_);

    // ---- concat([ctx,w]) @ w_np as two accumulating GEMMs + LN ----
    gemm_bf16_wmma<<<gU, 128, 0, stream>>>(hbuf, w_np, b_np, nullptr, xbuf, ML, U_, U_, 0);
    gemm_bf16_wmma<<<gU, 128, 0, stream>>>(wbuf, wnp_pad, nullptr, xbuf, xbuf, ML, U_, KPAD_, 0);
    layernorm_kernel<<<ML, 256, 0, stream>>>(xbuf, ln_np_g, ln_np_b, xbuf, U_);

    // ---- time-embedding MLP (silu), broadcast add ----
    temb_raw_kernel<<<64, 256, 0, stream>>>(times, traw);
    dim3 gT(U_ / 128, 1);
    gemm_bf16_wmma<<<gT, 128, 0, stream>>>(traw, t_w1, t_b1, nullptr, thid, 64, U_, U_, 2);
    gemm_bf16_wmma<<<gT, 128, 0, stream>>>(thid, t_w2, t_b2, nullptr, temb, 64, U_, U_, 0);
    add_temb_kernel<<<(B_ * L_ * U_ + 255) / 256, 256, 0, stream>>>(xbuf, temb);

    // ---- transformer layers ----
    for (int lyr = 0; lyr < DEPTH_; ++lyr) {
        const int lb = layer_base + lyr * 12;
        const float* ln1_g = (const float*)d_in[lb + 0];
        const float* ln1_b = (const float*)d_in[lb + 1];
        const float* wqkv  = (const float*)d_in[lb + 2];
        const float* bqkv  = (const float*)d_in[lb + 3];
        const float* wo    = (const float*)d_in[lb + 4];
        const float* bo    = (const float*)d_in[lb + 5];
        const float* ln2_g = (const float*)d_in[lb + 6];
        const float* ln2_b = (const float*)d_in[lb + 7];
        const float* w1    = (const float*)d_in[lb + 8];
        const float* b1    = (const float*)d_in[lb + 9];
        const float* w2    = (const float*)d_in[lb + 10];
        const float* b2    = (const float*)d_in[lb + 11];

        layernorm_kernel<<<ML, 256, 0, stream>>>(xbuf, ln1_g, ln1_b, hbuf, U_);
        dim3 gQ((3 * U_) / 128, ML / 64);
        gemm_bf16_wmma<<<gQ, 128, 0, stream>>>(hbuf, wqkv, bqkv, nullptr, qkvbuf, ML, 3 * U_, U_, 0);
        attn_kernel<<<B_ * HEADS_ * (L_ / 128), 256, 0, stream>>>(qkvbuf, validk, obuf);
        gemm_bf16_wmma<<<gU, 128, 0, stream>>>(obuf, wo, bo, xbuf, xbuf, ML, U_, U_, 0);  // residual

        layernorm_kernel<<<ML, 256, 0, stream>>>(xbuf, ln2_g, ln2_b, hbuf, U_);
        dim3 gF(FF_ / 128, ML / 64);
        gemm_bf16_wmma<<<gF, 128, 0, stream>>>(hbuf, w1, b1, nullptr, ffbuf, ML, FF_, U_, 1);   // gelu
        gemm_bf16_wmma<<<gU, 128, 0, stream>>>(ffbuf, w2, b2, xbuf, xbuf, ML, U_, FF_, 0);     // residual
    }

    // ---- final LN + output head (N padded to 128) + masked MSE ----
    layernorm_kernel<<<ML, 256, 0, stream>>>(xbuf, ln_f_g, ln_f_b, hbuf, U_);
    dim3 gO(KPAD_ / 128, ML / 64);
    gemm_bf16_wmma<<<gO, 128, 0, stream>>>(hbuf, wout_pad, bout_pad, nullptr, outbuf, ML, KPAD_, U_, 0);

    loss_partial_kernel<<<nparts, 256, 0, stream>>>(outbuf, vtbuf, validk, partial);
    loss_final_kernel<<<1, 256, 0, stream>>>(partial, nparts, validk, (float*)d_out);
}
